// FPModule_4209067950299
// MI455X (gfx1250) — compile-verified, matrix operand order checked by reading, not run on hardware
//
#include <hip/hip_runtime.h>
#include <hip/hip_bf16.h>

// ---------------------------------------------------------------------------
// Problem constants (from reference)
// ---------------------------------------------------------------------------
#define BB     4
#define NPTS   2048      // source points per batch
#define MPTS   8192      // query points per batch
#define KNN    3
#define C_IN   256
#define C_SKIP 128
#define C_CAT  384       // C_IN + C_SKIP
#define HID    256
#define C_OUT  128
#define NQ     (BB * MPTS)   // 32768 total queries

typedef __attribute__((ext_vector_type(16))) _Float16 v16h;
typedef __attribute__((ext_vector_type(8)))  _Float16 v8h;
typedef __attribute__((ext_vector_type(8)))  float    v8f;

// Workspace layout (bytes)
#define WS_A1    0                                   // [NQ x 384] f16 concat input
#define WS_H     (WS_A1 + (size_t)NQ * C_CAT * 2)    // [NQ x 256] f16 hidden
#define WS_W1T   (WS_H  + (size_t)NQ * HID   * 2)    // [256 x 384] f16 (K-contig)
#define WS_W2T   (WS_W1T + (size_t)HID * C_CAT * 2)  // [128 x 256] f16 (K-contig)

// ---------------------------------------------------------------------------
// Kernel 0: convert + transpose weights to f16 with K contiguous per out-row
// ---------------------------------------------------------------------------
__global__ void prep_weights_kernel(const float* __restrict__ W1,
                                    const float* __restrict__ W2,
                                    _Float16* __restrict__ W1T,
                                    _Float16* __restrict__ W2T) {
    int t = blockIdx.x * blockDim.x + threadIdx.x;
    if (t < HID * C_CAT) {                 // W1T[n][k] = W1[k][n]
        int n = t / C_CAT, k = t % C_CAT;
        W1T[t] = (_Float16)W1[k * HID + n];
    } else if (t < HID * C_CAT + C_OUT * HID) {  // W2T[n][k] = W2[k][n]
        int t2 = t - HID * C_CAT;
        int n = t2 / HID, k = t2 % HID;
        W2T[t2] = (_Float16)W2[k * C_OUT + n];
    }
}

// ---------------------------------------------------------------------------
// Kernel 1: fused 3-NN inverse-distance interpolation + concat, f16 output.
// One wave32 per query point; 8 waves (8 queries) per block; per-batch source
// positions staged in LDS (24 KB).
// ---------------------------------------------------------------------------
__device__ __forceinline__ void top3_insert(float d, int i,
                                            float& d0, float& d1, float& d2,
                                            int& i0, int& i1, int& i2) {
    if (d < d0)      { d2 = d1; i2 = i1; d1 = d0; i1 = i0; d0 = d; i0 = i; }
    else if (d < d1) { d2 = d1; i2 = i1; d1 = d;  i1 = i; }
    else if (d < d2) { d2 = d;  i2 = i; }
}

__global__ void __launch_bounds__(256)
knn_interp_kernel(const float* __restrict__ x,
                  const float* __restrict__ pos,
                  const float* __restrict__ x_skip,
                  const float* __restrict__ pos_skip,
                  _Float16* __restrict__ A1) {
    __shared__ float s_pos[NPTS * 3];

    const int lane = threadIdx.x & 31;
    const int qid  = blockIdx.x * 8 + (threadIdx.x >> 5);   // query id
    const int b    = qid >> 13;                             // qid / MPTS

    // cooperative stage of this batch's source positions into LDS
    const float* pb = pos + (size_t)b * NPTS * 3;
    for (int i = threadIdx.x; i < NPTS * 3; i += blockDim.x) s_pos[i] = pb[i];
    __syncthreads();

    const float qx = pos_skip[qid * 3 + 0];
    const float qy = pos_skip[qid * 3 + 1];
    const float qz = pos_skip[qid * 3 + 2];

    // per-lane top-3 over a strided slice of the 2048 candidates
    float d0 = 3.4e38f, d1 = 3.4e38f, d2 = 3.4e38f;
    int   i0 = 0, i1 = 0, i2 = 0;
    for (int n = lane; n < NPTS; n += 32) {
        float dx = qx - s_pos[n * 3 + 0];
        float dy = qy - s_pos[n * 3 + 1];
        float dz = qz - s_pos[n * 3 + 2];
        float dd = dx * dx + dy * dy + dz * dz;
        top3_insert(dd, n, d0, d1, d2, i0, i1, i2);
    }
    // wave32 butterfly merge of sorted triples
    #pragma unroll
    for (int m = 16; m >= 1; m >>= 1) {
        float e0 = __shfl_xor(d0, m, 32), e1 = __shfl_xor(d1, m, 32), e2 = __shfl_xor(d2, m, 32);
        int   j0 = __shfl_xor(i0, m, 32), j1 = __shfl_xor(i1, m, 32), j2 = __shfl_xor(i2, m, 32);
        top3_insert(e0, j0, d0, d1, d2, i0, i1, i2);
        top3_insert(e1, j1, d0, d1, d2, i0, i1, i2);
        top3_insert(e2, j2, d0, d1, d2, i0, i1, i2);
    }

    const float w0 = 1.0f / (fmaxf(d0, 0.0f) + 1e-16f);
    const float w1 = 1.0f / (fmaxf(d1, 0.0f) + 1e-16f);
    const float w2 = 1.0f / (fmaxf(d2, 0.0f) + 1e-16f);
    const float inv = 1.0f / (w0 + w1 + w2);
    const float a0 = w0 * inv, a1 = w1 * inv, a2 = w2 * inv;

    const float* f0 = x + ((size_t)(b * NPTS + i0)) * C_IN;
    const float* f1 = x + ((size_t)(b * NPTS + i1)) * C_IN;
    const float* f2 = x + ((size_t)(b * NPTS + i2)) * C_IN;
    _Float16* arow = A1 + (size_t)qid * C_CAT;

    #pragma unroll
    for (int c = lane; c < C_IN; c += 32)
        arow[c] = (_Float16)(a0 * f0[c] + a1 * f1[c] + a2 * f2[c]);
    const float* xs = x_skip + (size_t)qid * C_SKIP;
    #pragma unroll
    for (int c = lane; c < C_SKIP; c += 32)
        arow[C_IN + c] = (_Float16)xs[c];
}

// ---------------------------------------------------------------------------
// Kernel 2: WMMA GEMM  C[M x N] = relu?(A[M x K] * BT[N x K]^T + bias)
// Each wave computes a 16x64 output strip (4 N-tiles). Per K-step: all 10
// b128 fragment loads (2 A + 8 B) are issued together, then the 4
// v_wmma_f32_16x16x32_f16 run as a back-to-back group with the same A
// operand, letting the scheduler retire loads under the matrix pipe.
// ---------------------------------------------------------------------------
template <int KDIM, int NDIM, bool RELU, bool HALF_OUT>
__global__ void __launch_bounds__(256)
gemm_wmma_kernel(const _Float16* __restrict__ A,
                 const _Float16* __restrict__ BT,
                 const float* __restrict__ bias,
                 void* __restrict__ Cout) {
    constexpr int NTPW   = 4;                 // N-tiles per wave (16x64 strip)
    constexpr int GROUPS = NDIM / (16 * NTPW);
    const int wave = (blockIdx.x * blockDim.x + threadIdx.x) >> 5;
    const int lane = threadIdx.x & 31;
    const int tM  = (wave / GROUPS) * 16;
    const int tN0 = (wave % GROUPS) * (16 * NTPW);

    const int l15   = lane & 15;
    const int hi    = lane >> 4;      // 0 / 1
    const int kband = hi << 3;        // 0 / 8 : ISA f16 fragment K banding

    const _Float16* arow = A + (size_t)(tM + l15) * KDIM;
    const _Float16* brow[NTPW];
    #pragma unroll
    for (int j = 0; j < NTPW; ++j)
        brow[j] = BT + (size_t)(tN0 + j * 16 + l15) * KDIM;

    v8f acc[NTPW];
    #pragma unroll
    for (int j = 0; j < NTPW; ++j) acc[j] = (v8f){};

    #pragma unroll
    for (int kk = 0; kk < KDIM; kk += 32) {
        // ---- issue every fragment load for this K-step up front ----
        union { v16h v; v8h h[2]; } af, bf[NTPW];
        af.h[0] = *(const v8h*)(arow + kk + kband);        // K = kk+kband .. +7
        af.h[1] = *(const v8h*)(arow + kk + kband + 16);   // K = kk+kband+16 .. +23
        #pragma unroll
        for (int j = 0; j < NTPW; ++j) {
            bf[j].h[0] = *(const v8h*)(brow[j] + kk + kband);
            bf[j].h[1] = *(const v8h*)(brow[j] + kk + kband + 16);
        }
        if (kk + 32 < KDIM)                                // global_prefetch_b8
            __builtin_prefetch((const void*)(arow + kk + 32), 0, 1);
        // ---- back-to-back WMMA group sharing the A operand ----
        #pragma unroll
        for (int j = 0; j < NTPW; ++j) {
            acc[j] = __builtin_amdgcn_wmma_f32_16x16x32_f16(
                /*neg_a=*/false, af.v, /*neg_b=*/false, bf[j].v,
                /*c_mod=*/(short)0, acc[j], /*reuse_a=*/false, /*reuse_b=*/false);
        }
    }

    #pragma unroll
    for (int j = 0; j < NTPW; ++j) {
        const int col = tN0 + j * 16 + l15;
        const float bv = bias[col];
        #pragma unroll
        for (int r = 0; r < 8; ++r) {
            const int row = tM + r + (hi << 3);            // ISA C/D layout
            float v = acc[j][r] + bv;
            if (RELU) v = fmaxf(v, 0.0f);
            if (HALF_OUT)
                ((_Float16*)Cout)[(size_t)row * NDIM + col] = (_Float16)v;
            else
                ((float*)Cout)[(size_t)row * NDIM + col] = v;
        }
    }
}

// ---------------------------------------------------------------------------
// Kernel 3: pass-through tail (pos_skip f32, batch_skip int64)
// ---------------------------------------------------------------------------
__global__ void tail_copy_kernel(const float* __restrict__ pos_skip,
                                 const long long* __restrict__ batch_skip,
                                 float* __restrict__ out_f32) {
    int t = blockIdx.x * blockDim.x + threadIdx.x;
    float* pos_dst = out_f32 + (size_t)NQ * C_OUT;
    if (t < NQ * 3) {
        pos_dst[t] = pos_skip[t];
    } else if (t < NQ * 3 + NQ) {
        int j = t - NQ * 3;
        long long* b_dst = (long long*)(pos_dst + NQ * 3);
        b_dst[j] = batch_skip[j];
    }
}

// ---------------------------------------------------------------------------
// Launch
// ---------------------------------------------------------------------------
extern "C" void kernel_launch(void* const* d_in, const int* in_sizes, int n_in,
                              void* d_out, int out_size, void* d_ws, size_t ws_size,
                              hipStream_t stream) {
    const float* x         = (const float*)d_in[0];
    const float* pos       = (const float*)d_in[1];
    const float* x_skip    = (const float*)d_in[2];
    const float* pos_skip  = (const float*)d_in[3];
    const float* W1        = (const float*)d_in[4];
    const float* b1        = (const float*)d_in[5];
    const float* W2        = (const float*)d_in[6];
    const float* b2        = (const float*)d_in[7];
    const long long* batch_skip = (const long long*)d_in[9];

    char* ws = (char*)d_ws;
    _Float16* A1  = (_Float16*)(ws + WS_A1);
    _Float16* Hb  = (_Float16*)(ws + WS_H);
    _Float16* W1T = (_Float16*)(ws + WS_W1T);
    _Float16* W2T = (_Float16*)(ws + WS_W2T);
    float* out = (float*)d_out;

    // 0) weights -> f16, K-contiguous
    {
        int total = HID * C_CAT + C_OUT * HID;          // 131072
        prep_weights_kernel<<<(total + 255) / 256, 256, 0, stream>>>(W1, W2, W1T, W2T);
    }
    // 1) fused knn-interp + concat -> A1 [NQ x 384] f16
    knn_interp_kernel<<<NQ / 8, 256, 0, stream>>>(x, pos, x_skip, pos_skip, A1);

    // 2) H = relu(A1 @ W1 + b1) -> f16 [NQ x 256]
    {
        int waves = (NQ / 16) * (HID / 64);             // 8192 waves (16x64 strips)
        gemm_wmma_kernel<C_CAT, HID, true, true>
            <<<waves * 32 / 256, 256, 0, stream>>>(A1, W1T, b1, (void*)Hb);
    }
    // 3) out = H @ W2 + b2 -> f32 [NQ x 128]
    {
        int waves = (NQ / 16) * (C_OUT / 64);           // 4096 waves
        gemm_wmma_kernel<HID, C_OUT, false, false>
            <<<waves * 32 / 256, 256, 0, stream>>>(Hb, W2T, b2, (void*)out);
    }
    // 4) tail: pos_skip + batch_skip pass-through
    {
        int total = NQ * 3 + NQ;                        // 131072
        tail_copy_kernel<<<(total + 255) / 256, 256, 0, stream>>>(pos_skip, batch_skip, out);
    }
}
